// InteractionTransformerWithGAT_58411555225854
// MI455X (gfx1250) — matrix operand order repaired and checked by reading, not compiled, not run on hardware
//
#include <hip/hip_runtime.h>
#include <hip/hip_bf16.h>
#include <cstdint>

// ---------------------------------------------------------------------------
// Model constants (from reference)
// ---------------------------------------------------------------------------
#define NHEADS   8
#define HEADD    16
#define DMODEL   128
#define TSEQ     50          // HIST == NAG == 50
#define PREDN    60
#define NBATCH   64
#define KNN      8           // K

static constexpr int M_TIME = 64 * 50 * 50;   // 160000 rows (B*NAG, HIST) flattened
static constexpr int M_AG   = 64 * 50;        // 3200 rows  (B, NAG) flattened

typedef __attribute__((ext_vector_type(16))) _Float16 v16h;
typedef __attribute__((ext_vector_type(8)))  _Float16 v8h;
typedef __attribute__((ext_vector_type(8)))  float    v8f;

typedef unsigned int u32x4 __attribute__((ext_vector_type(4)));
typedef int          i32x4 __attribute__((ext_vector_type(4)));
typedef int          i32x8 __attribute__((ext_vector_type(8)));

// ---------------------------------------------------------------------------
// f32 -> f16 conversion (weights)
// ---------------------------------------------------------------------------
__global__ void k_f2h(const float* __restrict__ s, _Float16* __restrict__ d, int n) {
    int i = blockIdx.x * 256 + threadIdx.x;
    if (i < n) d[i] = (_Float16)s[i];
}

// ---------------------------------------------------------------------------
// WMMA GEMM:  out[M,N] = A[M,K] * W[N,K]^T (+bias)
// Block = 128 threads (4 waves), block tile = 64 rows x 32 cols.
// The 32 x Kd weight strip is staged into LDS once per block via the
// Tensor Data Mover (TDM), then each wave loads one A fragment per K-step
// and feeds two v_wmma_f32_16x16x32_f16 (two N tiles) from LDS.
// epi: 0 = f32 out, 1 = f16 out, 2 = relu -> f16 out
// NOTE: no static __shared__ here -> dynamic LDS starts at offset 0,
// which is what the TDM descriptor's lds_addr assumes.
// ---------------------------------------------------------------------------
__global__ void k_gemm(const _Float16* __restrict__ A, int lda,
                       const _Float16* __restrict__ W,
                       const float* __restrict__ bias,
                       void* __restrict__ out, int ldo,
                       int Kd, int epi)
{
    extern __shared__ _Float16 Bs[];            // 32 x Kd strip of W
    const int tn0 = blockIdx.y << 5;            // first of 32 columns

#if __has_builtin(__builtin_amdgcn_tensor_load_to_lds)
    if (threadIdx.x < 32) {                     // wave 0 issues the DMA
        unsigned long long ga =
            (unsigned long long)(uintptr_t)(W + (size_t)tn0 * Kd);
        // D# group 0: count=1, lds_addr=0, global_addr, type=2 ("image")
        u32x4 g0;
        g0[0] = 1u;                              // count=1, is_restore=0, gather off
        g0[1] = 0u;                              // lds_addr (bytes into our LDS)
        g0[2] = (unsigned)ga;                    // global_addr[31:0]
        g0[3] = ((unsigned)(ga >> 32) & 0x01FFFFFFu) | (2u << 30); // addr[56:32]|type=2
        // D# group 1: data_size=2B, tensor_dim0=Kd, tensor_dim1=32,
        //             tile_dim0=Kd, tile_dim1=32, tensor_dim0_stride=Kd
        i32x8 g1;
        g1[0] = (int)(1u << 16);                 // data_size=1 (2 bytes), wg_mask=0
        g1[1] = (int)((Kd & 0xFFFF) << 16);      // tensor_dim0[15:0] @ bits 63:48
        g1[2] = (int)(((unsigned)Kd >> 16) |     // tensor_dim0[31:16]
                      (32u << 16));              // tensor_dim1[15:0]
        g1[3] = (int)(((unsigned)Kd & 0xFFFF) << 16); // tensor_dim1[31:16]=0, tile_dim0=Kd
        g1[4] = 32;                              // tile_dim1=32, tile_dim2=0
        g1[5] = Kd;                              // tensor_dim0_stride[31:0]
        g1[6] = 0;                               // stride hi / dim1_stride lo
        g1[7] = 0;
        i32x4 gz = {0, 0, 0, 0};
#if defined(__clang_major__) && (__clang_major__ >= 23)
        i32x8 gz8 = {0, 0, 0, 0, 0, 0, 0, 0};
        __builtin_amdgcn_tensor_load_to_lds(g0, g1, gz, gz, gz8, 0);
#else
        __builtin_amdgcn_tensor_load_to_lds(g0, g1, gz, gz, 0);
#endif
        __builtin_amdgcn_s_wait_tensorcnt(0);
    }
#else
    {   // fallback: cooperative staging
        const v8h* src = (const v8h*)(W + (size_t)tn0 * Kd);
        v8h* dst = (v8h*)Bs;
        const int nv = (32 * Kd) >> 3;
        for (int i = threadIdx.x; i < nv; i += blockDim.x) dst[i] = src[i];
    }
#endif
    __syncthreads();

    const int lane = threadIdx.x & 31;
    const int wave = threadIdx.x >> 5;          // 0..3 -> M sub-tile
    const int tm   = (blockIdx.x << 6) + (wave << 4);

    // A fragment: lane<16 -> row=lane, K {k..k+15}
    //             lane>=16 -> row=lane-16, K {k+16..k+31}
    const int r    = lane & 15;
    const int koff = (lane >> 4) << 4;          // 0 or 16 halves
    const _Float16* Ap = A + (size_t)(tm + r) * lda + koff;

    // B fragments from LDS: lane holds column (local row of strip),
    // contiguous 16 K values; lanes>=16 take K range +16.
    const int kb = (lane >> 4) << 4;            // 0 or 16 halves
    const _Float16* B0 = Bs + (size_t)r * Kd + kb;          // cols tn0..tn0+15
    const _Float16* B1 = Bs + (size_t)(16 + r) * Kd + kb;   // cols tn0+16..tn0+31

    v8f acc0 = {}, acc1 = {};
    for (int k = 0; k < Kd; k += 32) {
        union { v16h v; v8h h[2]; } ua;
        ua.h[0] = *(const v8h*)(Ap + k);
        ua.h[1] = *(const v8h*)(Ap + k + 8);
        v16h b0 = *(const v16h*)(B0 + k);
        v16h b1 = *(const v16h*)(B1 + k);
        acc0 = __builtin_amdgcn_wmma_f32_16x16x32_f16(
                   false, ua.v, false, b0, (short)0, acc0, false, false);
        acc1 = __builtin_amdgcn_wmma_f32_16x16x32_f16(
                   false, ua.v, false, b1, (short)0, acc1, false, false);
    }

    // C/D layout: vgpr i, lane<16 -> (M=tm+i, N=col); lane>=16 -> (M=tm+8+i)
    const int mrow = tm + ((lane >> 4) << 3);
    #pragma unroll
    for (int ni = 0; ni < 2; ++ni) {
        const v8f acc = ni ? acc1 : acc0;
        const int col = tn0 + (ni << 4) + r;
        const float bv = bias ? bias[col] : 0.0f;
        if (epi == 0) {
            float* O = (float*)out;
            #pragma unroll
            for (int i = 0; i < 8; ++i)
                O[(size_t)(mrow + i) * ldo + col] = acc[i] + bv;
        } else if (epi == 1) {
            _Float16* O = (_Float16*)out;
            #pragma unroll
            for (int i = 0; i < 8; ++i)
                O[(size_t)(mrow + i) * ldo + col] = (_Float16)(acc[i] + bv);
        } else {
            _Float16* O = (_Float16*)out;
            #pragma unroll
            for (int i = 0; i < 8; ++i) {
                float v = acc[i] + bv;
                O[(size_t)(mrow + i) * ldo + col] = (_Float16)(v > 0.0f ? v : 0.0f);
            }
        }
    }
}

// ---------------------------------------------------------------------------
// Input projection: h[m,c] = x[m,:6] . in_proj_w[c,:] + b[c] + time_pos[t,c]
// ---------------------------------------------------------------------------
__global__ void k_in_proj(const float* __restrict__ x,
                          const float* __restrict__ W,   // 128 x 6
                          const float* __restrict__ b,
                          const float* __restrict__ tpos,// 50 x 128
                          float* __restrict__ h, _Float16* __restrict__ xA)
{
    size_t idx = (size_t)blockIdx.x * 256 + threadIdx.x;
    if (idx >= (size_t)M_TIME * DMODEL) return;
    int   c = idx & 127;
    size_t m = idx >> 7;
    int   t = (int)(m % TSEQ);
    const float* xp = x + m * 6;
    float acc = b[c] + tpos[t * DMODEL + c];
    #pragma unroll
    for (int d = 0; d < 6; ++d) acc += xp[d] * W[c * 6 + d];
    h[idx]  = acc;
    xA[idx] = (_Float16)acc;
}

// ---------------------------------------------------------------------------
// Attention: one block per (sequence, head). S = 50, head_dim = 16.
// ---------------------------------------------------------------------------
__global__ void k_attn(const _Float16* __restrict__ qkv,
                       _Float16* __restrict__ outA,
                       const int* __restrict__ mask)
{
    const int seq = blockIdx.x >> 3;
    const int hh  = blockIdx.x & 7;
    const _Float16* base = qkv + (size_t)seq * TSEQ * 384;

    __shared__ _Float16 Ks[TSEQ][HEADD];
    __shared__ _Float16 Vs[TSEQ][HEADD];
    __shared__ float    msk[TSEQ];

    const int tid = threadIdx.x;
    for (int i = tid; i < TSEQ * HEADD; i += blockDim.x) {
        int t = i >> 4, c = i & 15;
        Ks[t][c] = base[(size_t)t * 384 + 128 + hh * HEADD + c];
        Vs[t][c] = base[(size_t)t * 384 + 256 + hh * HEADD + c];
    }
    for (int t = tid; t < TSEQ; t += blockDim.x)
        msk[t] = (mask == nullptr || mask[seq * TSEQ + t]) ? 0.0f : -1e30f;
    __syncthreads();

    if (tid < TSEQ) {
        float q[HEADD];
        const _Float16* qp = base + (size_t)tid * 384 + hh * HEADD;
        #pragma unroll
        for (int c = 0; c < HEADD; ++c) q[c] = (float)qp[c];

        float s[TSEQ];
        float mx = -1e30f;
        for (int k = 0; k < TSEQ; ++k) {
            float d = 0.0f;
            #pragma unroll
            for (int c = 0; c < HEADD; ++c) d += q[c] * (float)Ks[k][c];
            d = d * 0.25f + msk[k];     // 1/sqrt(16)
            s[k] = d;
            mx = fmaxf(mx, d);
        }
        float den = 0.0f;
        float acc[HEADD];
        #pragma unroll
        for (int c = 0; c < HEADD; ++c) acc[c] = 0.0f;
        for (int k = 0; k < TSEQ; ++k) {
            float w = __expf(s[k] - mx);
            den += w;
            #pragma unroll
            for (int c = 0; c < HEADD; ++c) acc[c] += w * (float)Vs[k][c];
        }
        float inv = 1.0f / den;
        _Float16* op = outA + (size_t)(seq * TSEQ + tid) * DMODEL + hh * HEADD;
        #pragma unroll
        for (int c = 0; c < HEADD; ++c) op[c] = (_Float16)(acc[c] * inv);
    }
}

// ---------------------------------------------------------------------------
// Fused residual-add + LayerNorm over 128. One row per block (128 threads).
// ---------------------------------------------------------------------------
__global__ void k_add_ln(const float* __restrict__ resid,
                         const float* __restrict__ delta,
                         const float* __restrict__ g, const float* __restrict__ b,
                         float* __restrict__ outF, _Float16* __restrict__ out16)
{
    const int m = blockIdx.x, c = threadIdx.x;
    const size_t i = (size_t)m * DMODEL + c;
    float t = resid[i] + delta[i];

    __shared__ float red[DMODEL];
    red[c] = t;
    __syncthreads();
    for (int s2 = 64; s2 > 0; s2 >>= 1) {
        if (c < s2) red[c] += red[c + s2];
        __syncthreads();
    }
    float mu = red[0] * (1.0f / DMODEL);
    __syncthreads();
    float dv = t - mu;
    red[c] = dv * dv;
    __syncthreads();
    for (int s2 = 64; s2 > 0; s2 >>= 1) {
        if (c < s2) red[c] += red[c + s2];
        __syncthreads();
    }
    float var = red[0] * (1.0f / DMODEL);
    float yv = dv * rsqrtf(var + 1e-5f) * g[c] + b[c];
    outF[i]  = yv;
    out16[i] = (_Float16)yv;
}

// ---------------------------------------------------------------------------
// Gather last-timestep embedding to f16 (GAT GEMM input)
// ---------------------------------------------------------------------------
__global__ void k_gather(const float* __restrict__ h, _Float16* __restrict__ emb16)
{
    int idx = blockIdx.x * 256 + threadIdx.x;
    if (idx >= M_AG * DMODEL) return;
    int n = idx >> 7, c = idx & 127;
    emb16[idx] = (_Float16)h[((size_t)n * TSEQ + (TSEQ - 1)) * DMODEL + c];
}

// ---------------------------------------------------------------------------
// GAT: one block per batch. KNN(K+1), leaky-relu attention, segment softmax.
// ---------------------------------------------------------------------------
__global__ void k_gat(const float* __restrict__ hW,      // 3200 x 128
                      const float* __restrict__ x,       // original input (for pos)
                      const int* __restrict__ mask,
                      const float* __restrict__ att_src, // 8 x 16
                      const float* __restrict__ att_dst,
                      const float* __restrict__ gat_bias,
                      const float* __restrict__ agent_pos,
                      float* __restrict__ hA, _Float16* __restrict__ xA)
{
    const int bb = blockIdx.x;
    __shared__ float sh_hW[TSEQ][DMODEL];
    __shared__ float sas[TSEQ][NHEADS], sad[TSEQ][NHEADS];
    __shared__ float px[TSEQ], py[TSEQ];
    __shared__ unsigned long long adj[TSEQ];
    __shared__ int val[TSEQ];

    const int tid = threadIdx.x;
    for (int i = tid; i < TSEQ * DMODEL; i += blockDim.x)
        sh_hW[i >> 7][i & 127] = hW[(size_t)(bb * TSEQ) * DMODEL + i];
    for (int i = tid; i < TSEQ; i += blockDim.x) {
        size_t xi = (((size_t)(bb * TSEQ + i)) * TSEQ + (TSEQ - 1)) * 6;
        px[i] = x[xi];
        py[i] = x[xi + 1];
        val[i] = mask[bb * TSEQ + i];
    }
    __syncthreads();

    if (tid < TSEQ * NHEADS) {                 // a_src / a_dst
        int n = tid >> 3, hh2 = tid & 7;
        float s = 0.0f, d2 = 0.0f;
        #pragma unroll
        for (int c = 0; c < HEADD; ++c) {
            float v = sh_hW[n][hh2 * HEADD + c];
            s  += v * att_src[hh2 * HEADD + c];
            d2 += v * att_dst[hh2 * HEADD + c];
        }
        sas[n][hh2] = s;
        sad[n][hh2] = d2;
    }
    if (tid < TSEQ) {                          // KNN (K+1 incl. self), stable ties
        unsigned long long chosen = 0ull;
        for (int j = 0; j < KNN + 1; ++j) {
            float best = 3.4e38f;
            int bi = 0;
            for (int d = 0; d < TSEQ; ++d) {
                if ((chosen >> d) & 1ull) continue;
                float dx = px[tid] - px[d], dy = py[tid] - py[d];
                float dist = sqrtf(dx * dx + dy * dy);
                if (dist < best) { best = dist; bi = d; }
            }
            chosen |= 1ull << bi;
        }
        adj[tid] = chosen;
    }
    __syncthreads();

    if (tid < TSEQ * NHEADS) {
        const int d = tid >> 3, hh2 = tid & 7;
        float mx = -1e30f;
        for (int s = 0; s < TSEQ; ++s) {
            bool ex = (((adj[s] >> d) & 1ull) && val[s] && val[d] && s != d) || (s == d);
            if (!ex) continue;
            float e = sas[s][hh2] + sad[d][hh2];
            e = e > 0.0f ? e : 0.2f * e;
            mx = fmaxf(mx, e);
        }
        float den = 0.0f;
        float acc[HEADD];
        #pragma unroll
        for (int c = 0; c < HEADD; ++c) acc[c] = 0.0f;
        for (int s = 0; s < TSEQ; ++s) {
            bool ex = (((adj[s] >> d) & 1ull) && val[s] && val[d] && s != d) || (s == d);
            if (!ex) continue;
            float e = sas[s][hh2] + sad[d][hh2];
            e = e > 0.0f ? e : 0.2f * e;
            float w = __expf(e - mx);
            den += w;
            #pragma unroll
            for (int c = 0; c < HEADD; ++c) acc[c] += w * sh_hW[s][hh2 * HEADD + c];
        }
        float inv = 1.0f / den;
        #pragma unroll
        for (int c = 0; c < HEADD; ++c) {
            int cc = hh2 * HEADD + c;
            float o = acc[c] * inv + gat_bias[cc] + agent_pos[d * DMODEL + cc];
            size_t oi = (size_t)(bb * TSEQ + d) * DMODEL + cc;
            hA[oi] = o;
            xA[oi] = (_Float16)o;
        }
    }
}

// ---------------------------------------------------------------------------
// Final head: out[b, j] = a[b,0,:] . fc_w[j,:] + fc_b[j]    (j < 120)
// ---------------------------------------------------------------------------
__global__ void k_fc(const float* __restrict__ hA,
                     const float* __restrict__ W, const float* __restrict__ b,
                     float* __restrict__ out)
{
    const int bb = blockIdx.x, j = threadIdx.x;
    if (j >= PREDN * 2) return;
    const float* a = hA + (size_t)(bb * TSEQ) * DMODEL;
    float acc = b[j];
    for (int c = 0; c < DMODEL; ++c) acc += a[c] * W[j * DMODEL + c];
    out[bb * (PREDN * 2) + j] = acc;
}

// ---------------------------------------------------------------------------
// Host orchestration
// ---------------------------------------------------------------------------
extern "C" void kernel_launch(void* const* d_in, const int* in_sizes, int n_in,
                              void* d_out, int out_size, void* d_ws, size_t ws_size,
                              hipStream_t stream)
{
    auto F = [&](int i) { return (const float*)d_in[i]; };
    const float* X         = F(0);
    const int*   MASK      = (const int*)d_in[1];
    const float* in_proj_w = F(2);
    const float* in_proj_b = F(3);
    const float* time_pos  = F(4);
    auto TL = [&](int l, int f) { return F(5 + l * 12 + f); };   // time layers
    const float* lin_w     = F(53);
    const float* att_src   = F(54);
    const float* att_dst   = F(55);
    const float* gat_bias  = F(56);
    const float* agent_pos = F(57);
    auto AL = [&](int l, int f) { return F(58 + l * 12 + f); };  // agent layers
    const float* fc_w      = F(82);
    const float* fc_b      = F(83);

    // workspace layout
    char* wsb = (char*)d_ws;
    size_t off = 0;
    auto alloc = [&](size_t bytes) -> char* {
        char* p = wsb + off;
        off = (off + bytes + 255) & ~(size_t)255;
        return p;
    };
    float*    h     = (float*)   alloc((size_t)M_TIME * 128 * 4);
    _Float16* xA    = (_Float16*)alloc((size_t)M_TIME * 128 * 2);
    _Float16* xF    = (_Float16*)alloc((size_t)M_TIME * 512 * 2);
    _Float16* qkv   = (_Float16*)alloc((size_t)M_TIME * 384 * 2);
    float*    y     = (float*)   alloc((size_t)M_TIME * 128 * 4);
    _Float16* emb16 = (_Float16*)alloc((size_t)M_AG * 128 * 2);
    float*    hWb   = (float*)   alloc((size_t)M_AG * 128 * 4);

    auto cvt = [&](const float* src, size_t n) -> _Float16* {
        _Float16* dst = (_Float16*)alloc(n * 2);
        k_f2h<<<dim3((unsigned)((n + 255) / 256)), dim3(256), 0, stream>>>(src, dst, (int)n);
        return dst;
    };

    _Float16 *tw_in[4], *tw_out[4], *tw_w1[4], *tw_w2[4];
    for (int l = 0; l < 4; ++l) {
        tw_in[l]  = cvt(TL(l, 0), 384 * 128);
        tw_out[l] = cvt(TL(l, 2), 128 * 128);
        tw_w1[l]  = cvt(TL(l, 6), 512 * 128);
        tw_w2[l]  = cvt(TL(l, 8), 128 * 512);
    }
    _Float16 *aw_in[2], *aw_out[2], *aw_w1[2], *aw_w2[2];
    for (int l = 0; l < 2; ++l) {
        aw_in[l]  = cvt(AL(l, 0), 384 * 128);
        aw_out[l] = cvt(AL(l, 2), 128 * 128);
        aw_w1[l]  = cvt(AL(l, 6), 512 * 128);
        aw_w2[l]  = cvt(AL(l, 8), 128 * 512);
    }
    _Float16* lin16 = cvt(lin_w, 128 * 128);

    // block tile: 64 rows x 32 cols, 128 threads, LDS = 32*Kd f16 (TDM staged)
    auto gemm = [&](const _Float16* A, int lda, const _Float16* W, const float* bias,
                    void* out, int ldo, int M, int N, int Kd, int epi) {
        dim3 g(M / 64, N / 32);
        size_t sh = (size_t)32 * Kd * sizeof(_Float16);
        k_gemm<<<g, dim3(128), sh, stream>>>(A, lda, W, bias, out, ldo, Kd, epi);
    };

    auto enc = [&](int M, const _Float16* w_in, const float* b_in,
                   const _Float16* w_out, const float* b_out,
                   const float* g1, const float* bb1,
                   const _Float16* w1, const float* b1,
                   const _Float16* w2, const float* b2,
                   const float* g2, const float* bb2,
                   const int* mask) {
        gemm(xA, 128, w_in, b_in, qkv, 384, M, 384, 128, /*f16*/1);
        int nseq = M / TSEQ;
        k_attn<<<dim3(nseq * NHEADS), dim3(64), 0, stream>>>(qkv, xA, mask);
        gemm(xA, 128, w_out, b_out, y, 128, M, 128, 128, /*f32*/0);
        k_add_ln<<<dim3(M), dim3(128), 0, stream>>>(h, y, g1, bb1, h, xA);
        gemm(xA, 128, w1, b1, xF, 512, M, 512, 128, /*relu f16*/2);
        gemm(xF, 512, w2, b2, y, 128, M, 128, 512, /*f32*/0);
        k_add_ln<<<dim3(M), dim3(128), 0, stream>>>(h, y, g2, bb2, h, xA);
    };

    // 1) input projection + time positional embedding
    {
        size_t n = (size_t)M_TIME * DMODEL;
        k_in_proj<<<dim3((unsigned)((n + 255) / 256)), dim3(256), 0, stream>>>(
            X, in_proj_w, in_proj_b, time_pos, h, xA);
    }

    // 2) time transformer (4 layers, no mask)
    for (int l = 0; l < 4; ++l)
        enc(M_TIME, tw_in[l], TL(l, 1), tw_out[l], TL(l, 3), TL(l, 4), TL(l, 5),
            tw_w1[l], TL(l, 7), tw_w2[l], TL(l, 9), TL(l, 10), TL(l, 11), nullptr);

    // 3) gather last timestep, GAT linear (WMMA), GAT aggregation
    k_gather<<<dim3((M_AG * DMODEL + 255) / 256), dim3(256), 0, stream>>>(h, emb16);
    gemm(emb16, 128, lin16, nullptr, hWb, 128, M_AG, 128, 128, /*f32*/0);
    k_gat<<<dim3(NBATCH), dim3(512), 0, stream>>>(hWb, X, MASK, att_src, att_dst,
                                                  gat_bias, agent_pos, h, xA);

    // 4) agent transformer (2 layers, key mask)
    for (int l = 0; l < 2; ++l)
        enc(M_AG, aw_in[l], AL(l, 1), aw_out[l], AL(l, 3), AL(l, 4), AL(l, 5),
            aw_w1[l], AL(l, 7), aw_w2[l], AL(l, 9), AL(l, 10), AL(l, 11), MASK);

    // 5) final head
    k_fc<<<dim3(NBATCH), dim3(128), 0, stream>>>(h, fc_w, fc_b, (float*)d_out);
}
